// TransformerEncoderLayer_10307921511201
// MI455X (gfx1250) — matrix operand order repaired and checked by reading, not compiled, not run on hardware
//
#include <hip/hip_runtime.h>
#include <hip/hip_bf16.h>
#include <math.h>

// ---------------------------------------------------------------------------
// Transformer encoder layer (ISA block: long MHA -> short MHA -> FFN, 2x LN)
// for MI455X / gfx1250.  All GEMM-shaped work runs on v_wmma_f32_16x16x32_f16
// (f16 operands, f32 accumulate).  GEMM staging uses gfx1250 async-to-LDS
// copies (global_load_async_to_lds_b128 / s_wait_asynccnt) with double
// buffering, manually unrolled by two K-steps so buffer indices are
// compile-time constants (no cndmask address selects in the hot loop).
//
// Workspace requirement: ~292 MB (reused regions, see offsets below).
// ---------------------------------------------------------------------------

typedef _Float16     v16h  __attribute__((ext_vector_type(16)));
typedef float        v8f   __attribute__((ext_vector_type(8)));
typedef unsigned int u32x4 __attribute__((ext_vector_type(4)));

union FragH { v16h v; u32x4 u[2]; };   // 32B: one WMMA A/B fragment per lane
union AccF  { v8f  v; float f[8]; };   // 32B: one WMMA C/D fragment per lane

__device__ __forceinline__ v8f wmma_f16(v16h a, v16h b, v8f c) {
  // D = A(16x32 f16) * B(32x16 f16) + C(16x16 f32)
  return __builtin_amdgcn_wmma_f32_16x16x32_f16(false, a, false, b, (short)0, c,
                                                false, false);
}

// Raw 32-bit LDS offset of a generic pointer into a __shared__ object
// (addrspace(3) cast -> ptrtoint), as required by async-to-LDS VDST.
__device__ __forceinline__ unsigned lds_u32(const void* p) {
  typedef __attribute__((address_space(3))) const void lds_cv;
  return (unsigned)(unsigned long long)(lds_cv*)p;
}

// gfx1250 async DMA: 16B per lane, global -> LDS, tracked by ASYNCcnt.
__device__ __forceinline__ void async_b128(unsigned lds_off, const void* g) {
  asm volatile("global_load_async_to_lds_b128 %0, %1, off"
               :: "v"(lds_off), "v"(g) : "memory");
}
#define WAIT_ASYNC(n) asm volatile("s_wait_asynccnt " #n ::: "memory")

// A-fragment (16x32 f16), LDS tile stored row-major [m][k], ldh = halfs/row.
// lanes 0-15: row m=lane, halfs k={0..7,16..23}; lanes 16-31: m=lane-16,
// halfs k={8..15,24..31}.  ldh must be a multiple of 8 (16B alignment).
__device__ __forceinline__ v16h frag_a(const _Float16* base, int ldh) {
  const int lane = threadIdx.x & 31;
  const _Float16* p = base + (lane & 15) * ldh + ((lane >> 4) << 3);
  FragH f;
  f.u[0] = *(const u32x4*)(p);
  f.u[1] = *(const u32x4*)(p + 16);
  return f.v;
}

// B-fragment (32x16 f16), LDS tile stored transposed [n][k] (k contiguous).
// lanes 0-15: col n=lane, k={0..7,16..23}; lanes 16-31: col n=lane-16, k={8..15,24..31}.
__device__ __forceinline__ v16h frag_b(const _Float16* base, int ldh) {
  const int lane = threadIdx.x & 31;
  const _Float16* p = base + (lane & 15) * ldh + ((lane >> 4) << 3);
  FragH f;
  f.u[0] = *(const u32x4*)(p);
  f.u[1] = *(const u32x4*)(p + 16);
  return f.v;
}

// ---------------------------------------------------------------------------
// Generic WMMA GEMM:  C[r, n] = sum_k A[row_map(r)][k] * W[n][k]  (+ epilogue)
// Tile: 128(M) x 128(N), K-step 32 (K must be a multiple of 64).
// 256 threads = 8 waves in 4x2; each wave computes a 32x64 block
// (2x4 WMMA tiles -> 8 WMMA per K-step).  Async double-buffered staging.
// ---------------------------------------------------------------------------
constexpr int MT  = 128;
constexpr int NTG = 128;
constexpr int KS  = 32;
constexpr int LDA = 40;   // halfs (80B rows -> 16B aligned)
constexpr int LDB = 40;

template <typename P>
__global__ __launch_bounds__(256) void gemm_wmma(P p) {
  __shared__ _Float16 As[2][MT * LDA];
  __shared__ _Float16 Bs[2][NTG * LDB];

  const int tid  = threadIdx.x;
  const int lane = tid & 31;
  const int wid  = tid >> 5;
  const int wm   = wid >> 1;               // 0..3 : 32-row strip
  const int wn   = wid & 1;                // 0..1 : 64-col strip
  const int m_blk = blockIdx.y * MT;
  const int n_blk = blockIdx.x * NTG;

  const int lrow = tid >> 2;               // 0..63
  const int lch  = (tid & 3) * 8;          // half offset within row (16B chunk)

  const _Float16* a0 = p.a_row(m_blk + lrow, n_blk) + lch;
  const _Float16* a1 = p.a_row(m_blk + 64 + lrow, n_blk) + lch;
  const _Float16* b0 = p.W + (size_t)(n_blk + lrow) * p.K + lch;
  const _Float16* b1 = p.W + (size_t)(n_blk + 64 + lrow) * p.K + lch;

  unsigned sA0[2], sA1[2], sB0[2], sB1[2];
#pragma unroll
  for (int b = 0; b < 2; b++) {
    sA0[b] = lds_u32(&As[b][lrow * LDA + lch]);
    sA1[b] = lds_u32(&As[b][(64 + lrow) * LDA + lch]);
    sB0[b] = lds_u32(&Bs[b][lrow * LDB + lch]);
    sB1[b] = lds_u32(&Bs[b][(64 + lrow) * LDB + lch]);
  }

  AccF acc[2][4] = {};

  auto stage = [&](int b, int k) {
    async_b128(sA0[b], a0 + k); async_b128(sA1[b], a1 + k);
    async_b128(sB0[b], b0 + k); async_b128(sB1[b], b1 + k);
  };
  auto compute = [&](int b) {
    const v16h af0 = frag_a(&As[b][(wm * 32) * LDA], LDA);
    const v16h af1 = frag_a(&As[b][(wm * 32 + 16) * LDA], LDA);
#pragma unroll
    for (int j = 0; j < 4; j++) {
      const v16h bf = frag_b(&Bs[b][(wn * 64 + j * 16) * LDB], LDB);
      acc[0][j].v = wmma_f16(af0, bf, acc[0][j].v);
      acc[1][j].v = wmma_f16(af1, bf, acc[1][j].v);
    }
  };

  // software pipeline, unrolled by two K-steps (buffer index is constant)
  stage(0, 0);
  for (int k0 = 0; k0 < p.K; k0 += 2 * KS) {
    stage(1, k0 + KS);          // always valid: K % 64 == 0
    WAIT_ASYNC(4);              // buffer 0 (older 4 loads) has landed
    __syncthreads();
    compute(0);
    __syncthreads();            // all waves done with buffer 0

    if (k0 + 2 * KS < p.K) {
      stage(0, k0 + 2 * KS);
      WAIT_ASYNC(4);            // buffer 1 has landed
    } else {
      WAIT_ASYNC(0);
    }
    __syncthreads();
    compute(1);
    __syncthreads();            // all waves done with buffer 1
  }

  const int mrow = (lane >> 4) << 3;       // +8 rows for upper half-wave
#pragma unroll
  for (int i = 0; i < 2; i++)
#pragma unroll
    for (int j = 0; j < 4; j++)
#pragma unroll
      for (int r = 0; r < 8; r++) {
        const int m = m_blk + wm * 32 + i * 16 + r + mrow;
        const int n = n_blk + wn * 64 + j * 16 + (lane & 15);
        p.store(m, n, acc[i][j].f[r]);
      }
}

// ---------------------------------------------------------------------------
// Row-map / epilogue policies.
// Long-stage token order:  r = f_l = l*1024 + bt  (l in 64, bt in 1024)
// Short-stage token order: r = f_s = ls*512 + bs  (ls in 128, bs in 512)
// ---------------------------------------------------------------------------
struct PolQKVLong {  // Q,K from (src+pos), V from src; gathered via LONG->ORIG
  const _Float16 *xpos, *srch, *W; const float* bias; _Float16* out; int K;
  __device__ const _Float16* a_row(int r, int n0) const {
    const int l = r >> 10, bt = r & 1023;
    const int n = (l >> 3) * 1024 + ((bt >> 3) & 15) * 64 + (l & 7) * 8 + (bt & 7);
    const size_t fo = ((size_t)(bt >> 7) * 8192 + n) * 256;
    return (n0 < 512 ? xpos : srch) + fo;   // cols [0,512): x+pos ; [512,768): x
  }
  __device__ void store(int r, int c, float a) const {
    out[(size_t)r * 768 + c] = (_Float16)(a + bias[c]);
  }
};

struct PolOutLong {  // out-proj of long MHA; emits v (f16) and v+pos (f16)
  const _Float16 *A, *W; const float *bias, *pos; _Float16 *vsh, *vps; int K;
  __device__ const _Float16* a_row(int r, int) const { return A + (size_t)r * 256; }
  __device__ void store(int r, int c, float a) const {
    const float y = a + bias[c];
    vsh[(size_t)r * 256 + c] = (_Float16)y;
    const int l = r >> 10, bt = r & 1023;
    const int n = (l >> 3) * 1024 + ((bt >> 3) & 15) * 64 + (l & 7) * 8 + (bt & 7);
    vps[(size_t)r * 256 + c] = (_Float16)(y + pos[(size_t)n * 256 + c]);
  }
};

struct PolQKVShort { // gather rows via SHORT -> long-flat reinterpretation
  const _Float16 *vps, *vsh, *W; const float* bias; _Float16* out; int K;
  __device__ const _Float16* a_row(int r, int n0) const {
    const int ls = r >> 9, bs = r & 511;
    const int f = (bs >> 6) * 8192 + ((bs >> 3) & 7) * 1024 +
                  (ls >> 3) * 64 + (bs & 7) * 8 + (ls & 7);
    return (n0 < 512 ? vps : vsh) + (size_t)f * 256;
  }
  __device__ void store(int r, int c, float a) const {
    out[(size_t)r * 768 + c] = (_Float16)(a + bias[c]);
  }
};

struct PolOutShort { // out-proj of short MHA + residual, scattered to ORIG
  const _Float16 *A, *W; const float *bias, *src; float* X1; int K;
  __device__ const _Float16* a_row(int r, int) const { return A + (size_t)r * 256; }
  __device__ void store(int r, int c, float a) const {
    const int ls = r >> 9, bs = r & 511;
    const int n = (ls >> 3) * 512 + ((bs >> 3) & 7) * 64 + (ls & 7) * 8 + (bs & 7);
    const size_t o = ((size_t)(bs >> 6) * 8192 + n) * 256 + c;
    X1[o] = src[o] + a + bias[c];
  }
};

struct PolFFN1 {     // x @ w1^T + b1, ReLU -> f16 hidden
  const _Float16 *A, *W; const float* bias; _Float16* H; int K;
  __device__ const _Float16* a_row(int r, int) const { return A + (size_t)r * 256; }
  __device__ void store(int r, int c, float a) const {
    const float y = a + bias[c];
    H[(size_t)r * 1024 + c] = (_Float16)(y > 0.f ? y : 0.f);
  }
};

struct PolFFN2 {     // hidden @ w2^T + b2 + residual(xln)
  const _Float16 *A, *W; const float *bias, *xln; float* Z; int K;
  __device__ const _Float16* a_row(int r, int) const { return A + (size_t)r * 1024; }
  __device__ void store(int r, int c, float a) const {
    const size_t o = (size_t)r * 256 + c;
    Z[o] = xln[o] + a + bias[c];
  }
};

// ---------------------------------------------------------------------------
// Fused attention: one workgroup per (batch bt, head h).  L = seq len.
// Each wave owns a 16-row query strip; S=Q.K^T (one WMMA per 16x16 tile since
// Dh=32), softmax in registers with half-wave shfl reductions, O=P.V.
// Q/K staged with async-to-LDS; V staged transposed with normal LDS stores.
// ---------------------------------------------------------------------------
template <int L>
__global__ void attn_wmma(const _Float16* __restrict__ QKV,
                          _Float16* __restrict__ O, int BT, float scale) {
  constexpr int NT = L / 16;
  __shared__ _Float16 Qs[L * 32];
  __shared__ _Float16 Ks[L * 32];
  __shared__ _Float16 Vt[32 * L];  // transposed: [dh][l]
  __shared__ _Float16 Ps[L * L];

  const int bt = blockIdx.x, h = blockIdx.y;
  const int tid = threadIdx.x, lane = tid & 31, wid = tid >> 5;
  const int NTHR = L * 2;          // L/16 waves * 32

  for (int idx = tid; idx < L * 4; idx += NTHR) {
    const int l = idx >> 2, ch = (idx & 3) * 8;
    const _Float16* g = QKV + ((size_t)l * BT + bt) * 768 + h * 32 + ch;
    async_b128(lds_u32(&Qs[l * 32 + ch]), g);        // Q slice
    async_b128(lds_u32(&Ks[l * 32 + ch]), g + 256);  // K slice
  }
  for (int idx = tid; idx < L * 32; idx += NTHR) {
    const int l = idx >> 5, dh = idx & 31;
    Vt[dh * L + l] = QKV[((size_t)l * BT + bt) * 768 + 512 + h * 32 + dh];
  }
  WAIT_ASYNC(0);
  __syncthreads();

  const int m0 = wid * 16;
  const v16h qa = frag_a(&Qs[m0 * 32], 32);
  AccF s[NT];
#pragma unroll
  for (int j = 0; j < NT; j++) {
    v8f z = {};
    s[j].v = wmma_f16(qa, frag_b(&Ks[j * 16 * 32], 32), z);
  }

  // softmax over rows (row = lanes 0-15 or 16-31 of one C-VGPR, per tile col)
#pragma unroll
  for (int r = 0; r < 8; r++) {
    float mx = -1e30f;
#pragma unroll
    for (int j = 0; j < NT; j++) { s[j].f[r] *= scale; mx = fmaxf(mx, s[j].f[r]); }
    for (int m = 8; m; m >>= 1) mx = fmaxf(mx, __shfl_xor(mx, m));
    float sum = 0.f;
#pragma unroll
    for (int j = 0; j < NT; j++) {
      const float e = __expf(s[j].f[r] - mx);
      s[j].f[r] = e; sum += e;
    }
    for (int m = 8; m; m >>= 1) sum += __shfl_xor(sum, m);
    const float inv = 1.f / sum;
#pragma unroll
    for (int j = 0; j < NT; j++) s[j].f[r] *= inv;
  }

  const int mrow = (lane >> 4) << 3;
#pragma unroll
  for (int j = 0; j < NT; j++)
#pragma unroll
    for (int r = 0; r < 8; r++)
      Ps[(m0 + r + mrow) * L + j * 16 + (lane & 15)] = (_Float16)s[j].f[r];
  __syncthreads();

  AccF o[2] = {};
  for (int k0 = 0; k0 < L; k0 += 32) {
    const v16h pa = frag_a(&Ps[m0 * L + k0], L);
    o[0].v = wmma_f16(pa, frag_b(&Vt[0 * L + k0], L), o[0].v);
    o[1].v = wmma_f16(pa, frag_b(&Vt[16 * L + k0], L), o[1].v);
  }
#pragma unroll
  for (int j = 0; j < 2; j++)
#pragma unroll
    for (int r = 0; r < 8; r++) {
      const int m = m0 + r + mrow;
      O[((size_t)m * BT + bt) * 256 + h * 32 + j * 16 + (lane & 15)] =
          (_Float16)o[j].f[r];
    }
}

// ---------------------------------------------------------------------------
// Elementwise / LayerNorm kernels
// ---------------------------------------------------------------------------
__global__ void cvt16(const float* __restrict__ in, _Float16* __restrict__ out,
                      int n) {
  const int i = blockIdx.x * 256 + threadIdx.x;
  if (i < n) out[i] = (_Float16)in[i];
}

__global__ void prep_src_kernel(const float* __restrict__ src,
                                const float* __restrict__ pos,
                                _Float16* __restrict__ s16,
                                _Float16* __restrict__ x16) {
  const size_t i = (size_t)blockIdx.x * 256 + threadIdx.x;  // 8*8192*256 elems
  const float v = src[i];
  s16[i] = (_Float16)v;
  x16[i] = (_Float16)(v + pos[i & 2097151]);  // pos broadcast over batch
}

// LN over C=256, one wave per row (8 rows / 256-thread block)
__global__ __launch_bounds__(256) void ln_kernel(
    const float* __restrict__ X, const float* __restrict__ g,
    const float* __restrict__ be, _Float16* __restrict__ out16,
    float* __restrict__ outf) {
  const int row = blockIdx.x * 8 + (threadIdx.x >> 5);
  const int lane = threadIdx.x & 31;
  const float* xr = X + (size_t)row * 256;
  float v[8]; float s = 0.f;
#pragma unroll
  for (int i = 0; i < 8; i++) { v[i] = xr[lane + 32 * i]; s += v[i]; }
  for (int m = 16; m; m >>= 1) s += __shfl_xor(s, m);
  const float mean = s * 0.00390625f;
  float q = 0.f;
#pragma unroll
  for (int i = 0; i < 8; i++) { const float d = v[i] - mean; q += d * d; }
  for (int m = 16; m; m >>= 1) q += __shfl_xor(q, m);
  const float rstd = rsqrtf(q * 0.00390625f + 1e-5f);
#pragma unroll
  for (int i = 0; i < 8; i++) {
    const int c = lane + 32 * i;
    const float y = (v[i] - mean) * rstd * g[c] + be[c];
    out16[(size_t)row * 256 + c] = (_Float16)y;
    outf[(size_t)row * 256 + c] = y;
  }
}

// LN2 + final layout reinterpretation: out[b, :, :] = x[b].T flattened
__global__ __launch_bounds__(256) void ln_final_kernel(
    const float* __restrict__ Z, const float* __restrict__ g,
    const float* __restrict__ be, float* __restrict__ out) {
  const int row = blockIdx.x * 8 + (threadIdx.x >> 5);  // b*8192 + n
  const int lane = threadIdx.x & 31;
  const float* xr = Z + (size_t)row * 256;
  float v[8]; float s = 0.f;
#pragma unroll
  for (int i = 0; i < 8; i++) { v[i] = xr[lane + 32 * i]; s += v[i]; }
  for (int m = 16; m; m >>= 1) s += __shfl_xor(s, m);
  const float mean = s * 0.00390625f;
  float q = 0.f;
#pragma unroll
  for (int i = 0; i < 8; i++) { const float d = v[i] - mean; q += d * d; }
  for (int m = 16; m; m >>= 1) q += __shfl_xor(q, m);
  const float rstd = rsqrtf(q * 0.00390625f + 1e-5f);
  const int b = row >> 13, n = row & 8191;
#pragma unroll
  for (int i = 0; i < 8; i++) {
    const int c = lane + 32 * i;
    const float y = (v[i] - mean) * rstd * g[c] + be[c];
    out[(size_t)b * 2097152 + (size_t)c * 8192 + n] = y;
  }
}

// ---------------------------------------------------------------------------
// Host launch
// ---------------------------------------------------------------------------
extern "C" void kernel_launch(void* const* d_in, const int* in_sizes, int n_in,
                              void* d_out, int out_size, void* d_ws,
                              size_t ws_size, hipStream_t stream) {
  (void)in_sizes; (void)n_in; (void)out_size; (void)ws_size;
  const float* src     = (const float*)d_in[0];
  const float* pos     = (const float*)d_in[1];
  const float* w_in_l  = (const float*)d_in[2];
  const float* b_in_l  = (const float*)d_in[3];
  const float* w_out_l = (const float*)d_in[4];
  const float* b_out_l = (const float*)d_in[5];
  const float* w_in_s  = (const float*)d_in[6];
  const float* b_in_s  = (const float*)d_in[7];
  const float* w_out_s = (const float*)d_in[8];
  const float* b_out_s = (const float*)d_in[9];
  const float* w1  = (const float*)d_in[10];
  const float* b1  = (const float*)d_in[11];
  const float* w2  = (const float*)d_in[12];
  const float* b2  = (const float*)d_in[13];
  const float* g1  = (const float*)d_in[14];
  const float* be1 = (const float*)d_in[15];
  const float* g2  = (const float*)d_in[16];
  const float* be2 = (const float*)d_in[17];
  float* out = (float*)d_out;

  char* ws = (char*)d_ws;
  const size_t MB = (size_t)1 << 20;
  // region 0: f16 weights (2MB)
  _Float16* hW    = (_Float16*)ws;
  _Float16* hWinL = hW;             // 768x256
  _Float16* hWoL  = hW + 196608;    // 256x256
  _Float16* hWinS = hW + 262144;    // 768x256
  _Float16* hWoS  = hW + 458752;    // 256x256
  _Float16* hW1   = hW + 524288;    // 1024x256
  _Float16* hW2   = hW + 786432;    // 256x1024
  // region 1 [4MB,68MB): src16/xpos16 -> later X1/Z (f32)
  _Float16* hSrc  = (_Float16*)(ws + 4 * MB);
  _Float16* hXpos = (_Float16*)(ws + 36 * MB);
  float*    X1    = (float*)(ws + 4 * MB);
  // region 2 [68MB,196MB): QKV(long) -> QKV(short) -> FFN hidden
  _Float16* QKV   = (_Float16*)(ws + 68 * MB);   // [65536][768]
  _Float16* hHid  = (_Float16*)(ws + 68 * MB);   // [65536][1024]
  // region 3 [196MB,228MB): attn O (long/short) -> x16 after LN1
  _Float16* Obuf  = (_Float16*)(ws + 196 * MB);  // [65536][256]
  _Float16* hX16  = (_Float16*)(ws + 196 * MB);
  // region 4 [228MB,292MB): v+pos / v (f16) -> Xln (f32)
  _Float16* hVps  = (_Float16*)(ws + 228 * MB);
  _Float16* hVsh  = (_Float16*)(ws + 260 * MB);
  float*    Xln   = (float*)(ws + 228 * MB);

  const float scale = 0.17677669529663687f;  // 1/sqrt(32)

  // 1) weight conversion + input prep
  cvt16<<<768, 256, 0, stream>>>(w_in_l, hWinL, 196608);
  cvt16<<<256, 256, 0, stream>>>(w_out_l, hWoL, 65536);
  cvt16<<<768, 256, 0, stream>>>(w_in_s, hWinS, 196608);
  cvt16<<<256, 256, 0, stream>>>(w_out_s, hWoS, 65536);
  cvt16<<<1024, 256, 0, stream>>>(w1, hW1, 262144);
  cvt16<<<1024, 256, 0, stream>>>(w2, hW2, 262144);
  prep_src_kernel<<<65536, 256, 0, stream>>>(src, pos, hSrc, hXpos);

  // 2) long-range MHA
  { PolQKVLong p{hXpos, hSrc, hWinL, b_in_l, QKV, 256};
    gemm_wmma<<<dim3(6, 512), 256, 0, stream>>>(p); }
  attn_wmma<64><<<dim3(1024, 8), 128, 0, stream>>>(QKV, Obuf, 1024, scale);
  { PolOutLong p{Obuf, hWoL, b_out_l, pos, hVsh, hVps, 256};
    gemm_wmma<<<dim3(2, 512), 256, 0, stream>>>(p); }

  // 3) short-range MHA
  { PolQKVShort p{hVps, hVsh, hWinS, b_in_s, QKV, 256};
    gemm_wmma<<<dim3(6, 512), 256, 0, stream>>>(p); }
  attn_wmma<128><<<dim3(512, 8), 256, 0, stream>>>(QKV, Obuf, 512, scale);
  { PolOutShort p{Obuf, hWoS, b_out_s, src, X1, 256};
    gemm_wmma<<<dim3(2, 512), 256, 0, stream>>>(p); }

  // 4) LN1 + FFN + LN2 (+ final permute)
  ln_kernel<<<8192, 256, 0, stream>>>(X1, g1, be1, hX16, Xln);
  { PolFFN1 p{hX16, hW1, b1, hHid, 256};
    gemm_wmma<<<dim3(8, 512), 256, 0, stream>>>(p); }
  { PolFFN2 p{hHid, hW2, b2, Xln, X1, 1024};
    gemm_wmma<<<dim3(2, 512), 256, 0, stream>>>(p); }
  ln_final_kernel<<<8192, 256, 0, stream>>>(X1, g2, be2, out);
}
